// FPGA_GNN_GATv2_Enhanced_18511309046121
// MI455X (gfx1250) — matrix operand order repaired and checked by reading, not compiled, not run on hardware
//
#include <hip/hip_runtime.h>
#include <cstdint>
#include <cstddef>

// ---------------- constants from the reference ----------------
#define F_IN 128
#define HID  128
#define H_HEADS 4
#define HC   512           // HID*H
#define EDIM 32
#define TGT  4

// ---------------- types for WMMA ----------------
typedef __bf16 v16bf __attribute__((ext_vector_type(16)));
typedef float  v8f   __attribute__((ext_vector_type(8)));
typedef unsigned int v8u __attribute__((ext_vector_type(8)));

// ---------------- helpers ----------------
__device__ __forceinline__ unsigned short f2bf(float f) {
  unsigned int u = __float_as_uint(f);
  unsigned int r = u + 0x7FFFu + ((u >> 16) & 1u);   // round-to-nearest-even
  return (unsigned short)(r >> 16);
}

__device__ __forceinline__ void atomicMaxF(float* addr, float val) {
  if (val >= 0.f) atomicMax((int*)addr, __float_as_int(val));
  else            atomicMin((unsigned int*)addr, __float_as_uint(val));
}

// A fragment: 16x32 bf16, row-major source A[M x K] (lda = K)
__device__ __forceinline__ v16bf load_frag_a(const unsigned short* A, int lda,
                                             int row, int kb, int lane) {
  int base = (lane >> 4) << 3;                // lanes 0-15 -> 0, 16-31 -> 8
  const unsigned short* p = A + (size_t)row * lda + kb;
  v8u u;
#pragma unroll
  for (int j = 0; j < 8; ++j) {
    int k0 = (j < 4) ? (base + 2 * j) : (16 + base + 2 * (j - 4));
    u[j] = *(const unsigned int*)(p + k0);
  }
  return __builtin_bit_cast(v16bf, u);
}

// B fragment: 32x16 bf16 from pre-transposed weights Bt[K x N] (ldb = N)
// lane%16 -> N column, (vgpr,half) -> K row (same k order as A)
__device__ __forceinline__ v16bf load_frag_b(const unsigned short* Bt, int ldb,
                                             int kb, int n0, int lane) {
  int base = (lane >> 4) << 3;
  const unsigned short* p = Bt + (size_t)kb * ldb + n0 + (lane & 15);
  v8u u;
#pragma unroll
  for (int j = 0; j < 8; ++j) {
    int k0 = (j < 4) ? (base + 2 * j) : (16 + base + 2 * (j - 4));
    u[j] = (unsigned int)p[(size_t)k0 * ldb] | ((unsigned int)p[(size_t)(k0 + 1) * ldb] << 16);
  }
  return __builtin_bit_cast(v16bf, u);
}

// ---------------- WMMA GEMM, 2x2 register tiling ----------------
// C[MxN] = A[MxK](bf16) * Wt[KxN](bf16) + bias ; one wave -> 32x32 of C
// act: 0 = none, 1 = ELU.  Optional bf16 mirror output Cbf (ld = N).
__global__ void gemm_bf16_wmma(const unsigned short* __restrict__ A,
                               const unsigned short* __restrict__ Bt,
                               float* __restrict__ C,
                               unsigned short* __restrict__ Cbf,
                               const float* __restrict__ bias,
                               int M, int N, int K, int act) {
  int wave = (blockIdx.x * blockDim.x + threadIdx.x) >> 5;
  int lane = threadIdx.x & 31;
  int n2 = N >> 5;                       // 32-wide N tiles
  int m2 = (M + 31) >> 5;                // 32-tall M tiles
  if (wave >= m2 * n2) return;           // wave-uniform exit, EXEC stays all-1
  int mt = wave / n2, nt = wave % n2;
  int row0 = (mt << 5) + (lane & 15);
  int row1 = row0 + 16;
  if (row0 >= M) row0 = M - 1;           // clamp (no divergence)
  if (row1 >= M) row1 = M - 1;
  int n0 = nt << 5;
  v8f acc00 = {0.f,0.f,0.f,0.f,0.f,0.f,0.f,0.f};
  v8f acc01 = acc00, acc10 = acc00, acc11 = acc00;
  for (int kb = 0; kb < K; kb += 32) {
    v16bf a0 = load_frag_a(A, K, row0, kb, lane);
    v16bf a1 = load_frag_a(A, K, row1, kb, lane);
    v16bf b0 = load_frag_b(Bt, N, kb, n0, lane);
    v16bf b1 = load_frag_b(Bt, N, kb, n0 + 16, lane);
    acc00 = __builtin_amdgcn_wmma_f32_16x16x32_bf16(false, a0, false, b0, (short)0, acc00, false, false);
    acc01 = __builtin_amdgcn_wmma_f32_16x16x32_bf16(false, a0, false, b1, (short)0, acc01, false, false);
    acc10 = __builtin_amdgcn_wmma_f32_16x16x32_bf16(false, a1, false, b0, (short)0, acc10, false, false);
    acc11 = __builtin_amdgcn_wmma_f32_16x16x32_bf16(false, a1, false, b1, (short)0, acc11, false, false);
  }
  int nA = n0 + (lane & 15);
  int nB = nA + 16;
  int mbase = (mt << 5) + ((lane >> 4) << 3);
  float bvA = bias ? bias[nA] : 0.f;
  float bvB = bias ? bias[nB] : 0.f;
#pragma unroll
  for (int v = 0; v < 8; ++v) {
    int m0 = mbase + v;
    int m1 = m0 + 16;
    if (m0 < M) {
      float x0 = acc00[v] + bvA;
      float x1 = acc01[v] + bvB;
      if (act == 1) {
        x0 = (x0 > 0.f) ? x0 : (__expf(x0) - 1.f);
        x1 = (x1 > 0.f) ? x1 : (__expf(x1) - 1.f);
      }
      C[(size_t)m0 * N + nA] = x0;
      C[(size_t)m0 * N + nB] = x1;
      if (Cbf) { Cbf[(size_t)m0 * N + nA] = f2bf(x0); Cbf[(size_t)m0 * N + nB] = f2bf(x1); }
    }
    if (m1 < M) {
      float x0 = acc10[v] + bvA;
      float x1 = acc11[v] + bvB;
      if (act == 1) {
        x0 = (x0 > 0.f) ? x0 : (__expf(x0) - 1.f);
        x1 = (x1 > 0.f) ? x1 : (__expf(x1) - 1.f);
      }
      C[(size_t)m1 * N + nA] = x0;
      C[(size_t)m1 * N + nB] = x1;
      if (Cbf) { Cbf[(size_t)m1 * N + nA] = f2bf(x0); Cbf[(size_t)m1 * N + nB] = f2bf(x1); }
    }
  }
}

// ---------------- conversion kernels ----------------
__global__ void to_bf16_strided(const float* __restrict__ src,
                                unsigned short* __restrict__ dst,
                                int rows, int cols, int sld, int dld, int dcol) {
  long long total = (long long)rows * cols;
  long long stride = (long long)gridDim.x * blockDim.x;
  for (long long i = (long long)blockIdx.x * blockDim.x + threadIdx.x; i < total; i += stride) {
    int r = (int)(i / cols), c = (int)(i % cols);
    dst[(size_t)r * dld + dcol + c] = f2bf(src[(size_t)r * sld + c]);
  }
}

// W[Nout x K] (f32) -> Wt[K x Nout] (bf16)
__global__ void wt_to_bf16(const float* __restrict__ W, unsigned short* __restrict__ Wt,
                           int Nout, int K) {
  long long total = (long long)Nout * K;
  long long stride = (long long)gridDim.x * blockDim.x;
  for (long long i = (long long)blockIdx.x * blockDim.x + threadIdx.x; i < total; i += stride) {
    int n = (int)(i / K), k = (int)(i % K);
    Wt[(size_t)k * Nout + n] = f2bf(W[i]);
  }
}

__global__ void fill_neg_inf(float* p, int n) {
  int i = blockIdx.x * blockDim.x + threadIdx.x;
  if (i < n) p[i] = __int_as_float(0xFF800000);   // -inf
}

// ---------------- edge encoder: 1 -> 64(relu) -> 32 ----------------
__global__ void edge_encoder(const float* __restrict__ ea,
                             const float* __restrict__ w1, const float* __restrict__ b1,
                             const float* __restrict__ w2, const float* __restrict__ b2,
                             float* __restrict__ eout, int E) {
  int e = blockIdx.x * blockDim.x + threadIdx.x;
  if (e >= E) return;
  float a = ea[e];
  float h1[64];
#pragma unroll
  for (int j = 0; j < 64; ++j) h1[j] = fmaxf(w1[j] * a + b1[j], 0.f);
  for (int c = 0; c < 32; ++c) {
    float s = b2[c];
    const float* wr = w2 + c * 64;
#pragma unroll 8
    for (int j = 0; j < 64; ++j) s += wr[j] * h1[j];
    eout[(size_t)e * 32 + c] = s;
  }
}

// ---------------- GAT pass 1: per-edge logits (fused edge transform) ----------------
#define BLK1 256
__global__ void gat_logits(const float* __restrict__ xl, const float* __restrict__ xr,
                           const float* __restrict__ eemb,
                           const float* __restrict__ We,   // [512 x 32] row-major
                           const float* __restrict__ att,  // [4 x 128] flat = [512]
                           const int* __restrict__ src, const int* __restrict__ dst,
                           float* __restrict__ logits, float* __restrict__ lmax, int E) {
  __shared__ float es[BLK1 / 32][32];
  int lane = threadIdx.x & 31;
  int wid = threadIdx.x >> 5;
  int gw = blockIdx.x * (BLK1 / 32) + wid;
  int stride = gridDim.x * (BLK1 / 32);
  for (int e = gw; e < E; e += stride) {
    int s = src[e], d = dst[e];
    es[wid][lane] = eemb[(size_t)e * 32 + lane];
    float hs[4] = {0.f, 0.f, 0.f, 0.f};
#pragma unroll
    for (int t = 0; t < 16; ++t) {
      int c = t * 32 + lane;
      float acc = xl[(size_t)s * HC + c] + xr[(size_t)d * HC + c];
      const float* wrow = We + (size_t)c * 32;
#pragma unroll 8
      for (int k = 0; k < 32; ++k) acc += es[wid][k] * wrow[k];
      acc = (acc > 0.f) ? acc : 0.2f * acc;       // leaky_relu 0.2
      hs[t >> 2] += acc * att[c];                  // head*128 + (c%128) == c
    }
#pragma unroll
    for (int off = 16; off > 0; off >>= 1) {
      hs[0] += __shfl_xor(hs[0], off, 32);
      hs[1] += __shfl_xor(hs[1], off, 32);
      hs[2] += __shfl_xor(hs[2], off, 32);
      hs[3] += __shfl_xor(hs[3], off, 32);
    }
    if (lane == 0) {
#pragma unroll
      for (int h = 0; h < 4; ++h) {
        logits[(size_t)e * 4 + h] = hs[h];
        atomicMaxF(&lmax[(size_t)d * 4 + h], hs[h]);
      }
    }
  }
}

// ---------------- GAT pass 2: exp + denom ----------------
__global__ void gat_softmax_e(const float* __restrict__ logits, const float* __restrict__ lmax,
                              const int* __restrict__ dst,
                              float* __restrict__ alpha, float* __restrict__ denom, int E4) {
  int i = blockIdx.x * blockDim.x + threadIdx.x;
  if (i >= E4) return;
  int e = i >> 2, h = i & 3;
  int d = dst[e];
  float a = __expf(logits[i] - lmax[(size_t)d * 4 + h]);
  alpha[i] = a;
  atomicAdd(&denom[(size_t)d * 4 + h], a);
}

// ---------------- GAT pass 3: weighted scatter-add ----------------
__global__ void gat_aggregate(const float* __restrict__ xl,
                              const float* __restrict__ alpha, const float* __restrict__ denom,
                              const int* __restrict__ src, const int* __restrict__ dst,
                              float* __restrict__ agg) {
  int e = blockIdx.x;
  int c = threadIdx.x;                  // blockDim = 512
  int s = src[e], d = dst[e];
  int h = c >> 7;
  float w = alpha[(size_t)e * 4 + h] / (denom[(size_t)d * 4 + h] + 1e-16f);
  atomicAdd(&agg[(size_t)d * HC + c], xl[(size_t)s * HC + c] * w);
}

// ---------------- LayerNorm + ELU over 512 channels (wave per node) ----------------
__global__ void ln_elu(const float* __restrict__ in, const float* __restrict__ cbias,
                       const float* __restrict__ g, const float* __restrict__ b,
                       float* __restrict__ out, int N) {
  int lane = threadIdx.x & 31;
  int wid = threadIdx.x >> 5;
  int node = blockIdx.x * (blockDim.x >> 5) + wid;
  if (node >= N) return;
  float v[16];
  float s1 = 0.f, s2 = 0.f;
#pragma unroll
  for (int t = 0; t < 16; ++t) {
    int c = t * 32 + lane;
    float x = in[(size_t)node * HC + c] + cbias[c];
    v[t] = x; s1 += x; s2 += x * x;
  }
#pragma unroll
  for (int off = 16; off > 0; off >>= 1) {
    s1 += __shfl_xor(s1, off, 32);
    s2 += __shfl_xor(s2, off, 32);
  }
  float mu = s1 * (1.f / HC);
  float var = s2 * (1.f / HC) - mu * mu;
  float r = rsqrtf(var + 1e-5f);
#pragma unroll
  for (int t = 0; t < 16; ++t) {
    int c = t * 32 + lane;
    float y = (v[t] - mu) * r * g[c] + b[c];
    out[(size_t)node * HC + c] = (y > 0.f) ? y : (__expf(y) - 1.f);
  }
}

// ---------------- attention pooling head ----------------
__global__ void node_score(const float* __restrict__ hf, const float* __restrict__ gw,
                           const float* __restrict__ gb,
                           float* __restrict__ scores, float* __restrict__ ssum, int N) {
  int n = blockIdx.x * blockDim.x + threadIdx.x;
  if (n >= N) return;
  float s = gb[0];
#pragma unroll 8
  for (int k = 0; k < HID; ++k) s += hf[(size_t)n * HID + k] * gw[k];
  float v = __expf(tanhf(s));          // softmax numerator (tanh bounded, no max needed)
  scores[n] = v;
  atomicAdd(ssum, v);
}

__global__ void graph_pool(const float* __restrict__ hf, const float* __restrict__ scores,
                           const float* __restrict__ ssum, const int* __restrict__ batch,
                           float* __restrict__ ge) {
  int n = blockIdx.x;
  int c = threadIdx.x;                 // blockDim = 128
  float w = scores[n] / ssum[0];
  atomicAdd(&ge[(size_t)batch[n] * HID + c], hf[(size_t)n * HID + c] * w);
}

// ---------------- tiny MLP head (one thread per graph) ----------------
__global__ void mlp_head(const float* __restrict__ ge, const float* __restrict__ strat,
                         const float* __restrict__ w1, const float* __restrict__ b1,
                         const float* __restrict__ g1, const float* __restrict__ bb1,
                         const float* __restrict__ w2, const float* __restrict__ b2,
                         const float* __restrict__ g2, const float* __restrict__ bb2,
                         const float* __restrict__ w3, const float* __restrict__ b3,
                         float* __restrict__ out, int G) {
  int g = threadIdx.x;
  if (g >= G) return;
  float in[HID + 1];
  for (int k = 0; k < HID; ++k) in[k] = ge[(size_t)g * HID + k];
  in[HID] = strat[g];
  float m1[64];
  float mu = 0.f, vr = 0.f;
  for (int o = 0; o < 64; ++o) {
    float s = b1[o];
    const float* wr = w1 + o * (HID + 1);
    for (int k = 0; k < HID + 1; ++k) s += wr[k] * in[k];
    m1[o] = s; mu += s;
  }
  mu *= (1.f / 64.f);
  for (int o = 0; o < 64; ++o) { float d = m1[o] - mu; vr += d * d; }
  float r = rsqrtf(vr * (1.f / 64.f) + 1e-5f);
  for (int o = 0; o < 64; ++o) m1[o] = fmaxf((m1[o] - mu) * r * g1[o] + bb1[o], 0.f);
  float m2[32];
  mu = 0.f; vr = 0.f;
  for (int o = 0; o < 32; ++o) {
    float s = b2[o];
    const float* wr = w2 + o * 64;
    for (int k = 0; k < 64; ++k) s += wr[k] * m1[k];
    m2[o] = s; mu += s;
  }
  mu *= (1.f / 32.f);
  for (int o = 0; o < 32; ++o) { float d = m2[o] - mu; vr += d * d; }
  r = rsqrtf(vr * (1.f / 32.f) + 1e-5f);
  for (int o = 0; o < 32; ++o) m2[o] = fmaxf((m2[o] - mu) * r * g2[o] + bb2[o], 0.f);
  for (int o = 0; o < TGT; ++o) {
    float s = b3[o];
    const float* wr = w3 + o * 32;
    for (int k = 0; k < 32; ++k) s += wr[k] * m2[k];
    out[(size_t)g * TGT + o] = s;
  }
}

// ---------------- host orchestration ----------------
extern "C" void kernel_launch(void* const* d_in, const int* in_sizes, int n_in,
                              void* d_out, int out_size, void* d_ws, size_t ws_size,
                              hipStream_t stream) {
  const int N = in_sizes[0] / F_IN;        // 10000
  const int E = in_sizes[2];               // 160000
  const int G = in_sizes[4];               // 8

  // ---- input pointers (dict order, params flattened depth-first) ----
  const float* x      = (const float*)d_in[0];
  const int*   eidx   = (const int*)d_in[1];
  const int*   srcI   = eidx;
  const int*   dstI   = eidx + E;
  const float* eattr  = (const float*)d_in[2];
  const int*   batch  = (const int*)d_in[3];
  const float* strat  = (const float*)d_in[4];
  int pi = 5;
  const float* ee_w1 = (const float*)d_in[pi++];
  const float* ee_b1 = (const float*)d_in[pi++];
  const float* ee_w2 = (const float*)d_in[pi++];
  const float* ee_b2 = (const float*)d_in[pi++];
  struct ConvP { const float *lw, *lb, *rw, *rb, *ew, *att, *bias; } cp[3];
  for (int i = 0; i < 3; ++i) {
    cp[i].lw   = (const float*)d_in[pi++];
    cp[i].lb   = (const float*)d_in[pi++];
    cp[i].rw   = (const float*)d_in[pi++];
    cp[i].rb   = (const float*)d_in[pi++];
    cp[i].ew   = (const float*)d_in[pi++];
    cp[i].att  = (const float*)d_in[pi++];
    cp[i].bias = (const float*)d_in[pi++];
  }
  const float* ng[3]; const float* nb[3];
  for (int i = 0; i < 3; ++i) ng[i] = (const float*)d_in[pi++];
  for (int i = 0; i < 3; ++i) nb[i] = (const float*)d_in[pi++];
  const float* sw[2]; const float* sb[2];
  for (int i = 0; i < 2; ++i) { sw[i] = (const float*)d_in[pi++]; sb[i] = (const float*)d_in[pi++]; }
  const float* final_w = (const float*)d_in[pi++];
  const float* final_b = (const float*)d_in[pi++];
  const float* gatt_w  = (const float*)d_in[pi++];
  const float* gatt_b  = (const float*)d_in[pi++];
  const float* mw1 = (const float*)d_in[pi++];
  const float* mb1 = (const float*)d_in[pi++];
  const float* mg1 = (const float*)d_in[pi++];
  const float* mbb1 = (const float*)d_in[pi++];
  const float* mw2 = (const float*)d_in[pi++];
  const float* mb2 = (const float*)d_in[pi++];
  const float* mg2 = (const float*)d_in[pi++];
  const float* mbb2 = (const float*)d_in[pi++];
  const float* mw3 = (const float*)d_in[pi++];
  const float* mb3 = (const float*)d_in[pi++];

  // ---- workspace allocator ----
  size_t off = 0;
  auto alloc = [&](size_t bytes) -> void* {
    void* p = (char*)d_ws + off;
    off += (bytes + 255) & ~(size_t)255;
    return p;
  };
  float* xl    = (float*)alloc((size_t)N * HC * 4);
  float* xr    = (float*)alloc((size_t)N * HC * 4);
  float* agg   = (float*)alloc((size_t)N * HC * 4);
  float* hpost = (float*)alloc((size_t)N * HC * 4);
  float* h1    = (float*)alloc((size_t)N * HC * 4);
  float* h2    = (float*)alloc((size_t)N * HC * 4);
  float* h3    = (float*)alloc((size_t)N * HC * 4);
  float* eemb  = (float*)alloc((size_t)E * EDIM * 4);
  float* logits= (float*)alloc((size_t)E * 4 * 4);
  float* alpha = (float*)alloc((size_t)E * 4 * 4);
  float* lmax  = (float*)alloc((size_t)N * 4 * 4);
  float* denom = (float*)alloc((size_t)N * 4 * 4);
  float* hf    = (float*)alloc((size_t)N * HID * 4);
  float* scores= (float*)alloc((size_t)N * 4);
  float* ssum  = (float*)alloc(256);
  float* ge    = (float*)alloc((size_t)G * HID * 4);
  unsigned short* x_bf  = (unsigned short*)alloc((size_t)N * F_IN * 2);
  unsigned short* h1_bf = (unsigned short*)alloc((size_t)N * HC * 2);
  unsigned short* h2_bf = (unsigned short*)alloc((size_t)N * HC * 2);
  unsigned short* h3_bf = (unsigned short*)alloc((size_t)N * HC * 2);
  unsigned short* cat_bf = (unsigned short*)alloc((size_t)N * 1664 * 2);
  unsigned short* Wlt[3], *Wrt[3];
  int din_arr[3] = {F_IN, HC, HC};
  for (int i = 0; i < 3; ++i) {
    Wlt[i] = (unsigned short*)alloc((size_t)din_arr[i] * HC * 2);
    Wrt[i] = (unsigned short*)alloc((size_t)din_arr[i] * HC * 2);
  }
  unsigned short* skt0   = (unsigned short*)alloc((size_t)1152 * HC * 2);
  unsigned short* skt1   = (unsigned short*)alloc((size_t)1664 * HC * 2);
  unsigned short* fint   = (unsigned short*)alloc((size_t)HC * HID * 2);
  (void)ws_size; (void)n_in; (void)out_size;

  // ---- weight prep (bf16 transposed) ----
  for (int i = 0; i < 3; ++i) {
    wt_to_bf16<<<512, 256, 0, stream>>>(cp[i].lw, Wlt[i], HC, din_arr[i]);
    wt_to_bf16<<<512, 256, 0, stream>>>(cp[i].rw, Wrt[i], HC, din_arr[i]);
  }
  wt_to_bf16<<<1024, 256, 0, stream>>>(sw[0], skt0, HC, 1152);
  wt_to_bf16<<<1024, 256, 0, stream>>>(sw[1], skt1, HC, 1664);
  wt_to_bf16<<<256, 256, 0, stream>>>(final_w, fint, HID, HC);
  to_bf16_strided<<<1024, 256, 0, stream>>>(x, x_bf, N, F_IN, F_IN, F_IN, 0);

  // ---- edge encoder ----
  edge_encoder<<<(E + 255) / 256, 256, 0, stream>>>(eattr, ee_w1, ee_b1, ee_w2, ee_b2, eemb, E);

  auto gemm = [&](const unsigned short* A, const unsigned short* Bt, float* C,
                  unsigned short* Cbf, const float* bias, int M, int Nn, int K, int act) {
    int waves = ((M + 31) / 32) * (Nn / 32);
    int blocks = (waves + 7) / 8;
    gemm_bf16_wmma<<<blocks, 256, 0, stream>>>(A, Bt, C, Cbf, bias, M, Nn, K, act);
  };

  // ---- 3 GATv2 layers ----
  for (int i = 0; i < 3; ++i) {
    const unsigned short* hin = (i == 0) ? x_bf : (i == 1 ? h1_bf : h2_bf);
    int din = din_arr[i];
    gemm(hin, Wlt[i], xl, nullptr, cp[i].lb, N, HC, din, 0);
    gemm(hin, Wrt[i], xr, nullptr, cp[i].rb, N, HC, din, 0);

    fill_neg_inf<<<(N * 4 + 255) / 256, 256, 0, stream>>>(lmax, N * 4);
    hipMemsetAsync(denom, 0, (size_t)N * 4 * 4, stream);
    gat_logits<<<1024, BLK1, 0, stream>>>(xl, xr, eemb, cp[i].ew, cp[i].att,
                                          srcI, dstI, logits, lmax, E);
    gat_softmax_e<<<(E * 4 + 255) / 256, 256, 0, stream>>>(logits, lmax, dstI, alpha, denom, E * 4);
    hipMemsetAsync(agg, 0, (size_t)N * HC * 4, stream);
    gat_aggregate<<<E, HC, 0, stream>>>(xl, alpha, denom, srcI, dstI, agg);

    float* lnout = (i == 0) ? h1 : hpost;
    ln_elu<<<(N + 7) / 8, 256, 0, stream>>>(agg, cp[i].bias, ng[i], nb[i], lnout, N);

    if (i == 0) {
      to_bf16_strided<<<2048, 256, 0, stream>>>(h1, h1_bf, N, HC, HC, HC, 0);
    } else if (i == 1) {
      // concat [x | h1 | hpost], ld = 1152
      to_bf16_strided<<<1024, 256, 0, stream>>>(x, cat_bf, N, F_IN, F_IN, 1152, 0);
      to_bf16_strided<<<2048, 256, 0, stream>>>(h1, cat_bf, N, HC, HC, 1152, 128);
      to_bf16_strided<<<2048, 256, 0, stream>>>(hpost, cat_bf, N, HC, HC, 1152, 640);
      gemm(cat_bf, skt0, h2, h2_bf, sb[0], N, HC, 1152, 1);   // ELU fused, bf16 mirror
    } else {
      // concat [x | h1 | h2 | hpost], ld = 1664
      to_bf16_strided<<<1024, 256, 0, stream>>>(x, cat_bf, N, F_IN, F_IN, 1664, 0);
      to_bf16_strided<<<2048, 256, 0, stream>>>(h1, cat_bf, N, HC, HC, 1664, 128);
      to_bf16_strided<<<2048, 256, 0, stream>>>(h2, cat_bf, N, HC, HC, 1664, 640);
      to_bf16_strided<<<2048, 256, 0, stream>>>(hpost, cat_bf, N, HC, HC, 1664, 1152);
      gemm(cat_bf, skt1, h3, h3_bf, sb[1], N, HC, 1664, 1);   // ELU fused, bf16 mirror
    }
  }

  // ---- final projection + attention pooling ----
  gemm(h3_bf, fint, hf, nullptr, final_b, N, HID, HC, 0);
  hipMemsetAsync(ssum, 0, 256, stream);
  node_score<<<(N + 255) / 256, 256, 0, stream>>>(hf, gatt_w, gatt_b, scores, ssum, N);
  hipMemsetAsync(ge, 0, (size_t)G * HID * 4, stream);
  graph_pool<<<N, HID, 0, stream>>>(hf, scores, ssum, batch, ge);

  // ---- MLP head ----
  mlp_head<<<1, 32, 0, stream>>>(ge, strat, mw1, mb1, mg1, mbb1, mw2, mb2, mg2, mbb2,
                                 mw3, mb3, (float*)d_out, G);
}
